// SelfAttention_63221918597747
// MI455X (gfx1250) — compile-verified, matrix operand order checked by reading, not run on hardware
//
#include <hip/hip_runtime.h>
#include <hip/hip_bf16.h>

typedef __bf16 bf16_t;
typedef __attribute__((ext_vector_type(8)))  __bf16 v8bf;
typedef __attribute__((ext_vector_type(16))) __bf16 v16bf;
typedef __attribute__((ext_vector_type(8)))  float  v8f;

#define DIM   128
#define SEQ   4096
#define NB    8
#define LOG2E 1.4426950408889634f

static __device__ __forceinline__ v8bf ld8(const bf16_t* p) {
    return *reinterpret_cast<const v8bf*>(p);
}
static __device__ __forceinline__ v16bf cat8(v8bf lo, v8bf hi) {
    return __builtin_shufflevector(lo, hi, 0,1,2,3,4,5,6,7,8,9,10,11,12,13,14,15);
}

// ---------------------------------------------------------------------------
// Prep: fp32 -> bf16, row-major copy + [B,D,S] transposed copy (for V^T loads)
// ---------------------------------------------------------------------------
__global__ void cvt_x_kernel(const float* __restrict__ x,
                             bf16_t* __restrict__ xb,
                             bf16_t* __restrict__ xbT) {
    size_t i = (size_t)blockIdx.x * 256 + threadIdx.x;   // over B*S*D
    float v = x[i];
    bf16_t h = (bf16_t)v;
    xb[i] = h;
    int d = (int)(i & (DIM - 1));
    size_t bs = i >> 7;                // b*S + s
    int s = (int)(bs & (SEQ - 1));
    int b = (int)(bs >> 12);
    xbT[((size_t)(b * DIM + d) << 12) + s] = h;
}

__global__ void cvt_w_kernel(const float* __restrict__ W, bf16_t* __restrict__ Wb) {
    size_t i = (size_t)blockIdx.x * 256 + threadIdx.x;   // over D*2D
    Wb[i] = (bf16_t)W[i];
}

// ---------------------------------------------------------------------------
// One 32-key flash-attention step for a 16-query strip owned by one wave.
// MASK=false: interior step (k + 31 < qbase <= q for every lane) -> no compares.
// MASK=true : diagonal tail step (exactly one per wave).
// ---------------------------------------------------------------------------
template <bool MASK>
static __device__ __forceinline__ void attn_step32(
        const bf16_t* __restrict__ xrow,   // xb for this batch  [S, D]
        const bf16_t* __restrict__ xT,     // xbT for this batch [D, S]
        const v16bf (&bq)[4],              // Q^T B-operands, per 32-wide D chunk
        v8f (&acc)[8],                     // O^T accumulators (8 d-tiles)
        float& m_i, float& l_i,
        int kbase, int q, int half, int col) {

    const v8f zero = {0.f, 0.f, 0.f, 0.f, 0.f, 0.f, 0.f, 0.f};

    // ---- scores^T : two 16x16 tiles over keys [kbase, kbase+32) ----
    v8f sc0 = zero, sc1 = zero;
#pragma unroll
    for (int dch = 0; dch < 4; ++dch) {
        const int dlo = dch * 32 + half * 8;
        const bf16_t* k0r = xrow + (size_t)(kbase + col) * DIM + dlo;
        const bf16_t* k1r = xrow + (size_t)(kbase + 16 + col) * DIM + dlo;
        v16bf a0 = cat8(ld8(k0r), ld8(k0r + 16));
        v16bf a1 = cat8(ld8(k1r), ld8(k1r + 16));
        sc0 = __builtin_amdgcn_wmma_f32_16x16x32_bf16(
                  false, a0, false, bq[dch], (short)0, sc0, false, false);
        sc1 = __builtin_amdgcn_wmma_f32_16x16x32_bf16(
                  false, a1, false, bq[dch], (short)0, sc1, false, false);
    }

    // ---- (optional) causal mask + online softmax (keys along VGPRs) ----
    const int k0 = kbase + half * 8;
    const int k1 = k0 + 16;
    float s0[8], s1[8];
    float mt = -__builtin_inff();
#pragma unroll
    for (int v = 0; v < 8; ++v) {
        if (MASK) {
            s0[v] = (k0 + v <= q) ? sc0[v] : -__builtin_inff();
            s1[v] = (k1 + v <= q) ? sc1[v] : -__builtin_inff();
        } else {
            s0[v] = sc0[v];
            s1[v] = sc1[v];
        }
        mt = fmaxf(mt, fmaxf(s0[v], s1[v]));
    }
    mt = fmaxf(mt, __shfl_xor(mt, 16, 32));
    const float m_new = fmaxf(m_i, mt);
    const float alpha = exp2f((m_i - m_new) * LOG2E);

    float psum = 0.f;
    v16bf pb;   // exp(S^T) in C-layout == B-operand layout of P^T
#pragma unroll
    for (int v = 0; v < 8; ++v) {
        float p0 = exp2f((s0[v] - m_new) * LOG2E);
        float p1 = exp2f((s1[v] - m_new) * LOG2E);
        psum += p0 + p1;
        pb[v]     = (bf16_t)p0;   // elems 0..7  -> k = 8*half+j
        pb[v + 8] = (bf16_t)p1;   // elems 8..15 -> k = 16+8*half+j
    }
    psum += __shfl_xor(psum, 16, 32);
    l_i = l_i * alpha + psum;
    m_i = m_new;

    // ---- rescale accumulators, then O^T += V^T * P^T ----
#pragma unroll
    for (int t = 0; t < 8; ++t) {
#pragma unroll
        for (int v = 0; v < 8; ++v) acc[t][v] *= alpha;
    }
#pragma unroll
    for (int t = 0; t < 8; ++t) {
        const bf16_t* vr = xT + (size_t)(t * 16 + col) * SEQ + kbase + half * 8;
        v16bf av = cat8(ld8(vr), ld8(vr + 16));
        acc[t] = __builtin_amdgcn_wmma_f32_16x16x32_bf16(
                     false, av, false, pb, (short)0, acc[t], false, false);
    }
}

// ---------------------------------------------------------------------------
// Flash attention (causal, unscaled scores). One wave = 16 query rows.
// Computes S^T = K·Q^T so softmax is per-lane; C-layout of exp(S^T) is the
// B-operand layout of P^T for O^T = V^T·P^T  -> no cross-lane movement.
// ---------------------------------------------------------------------------
__global__ void __launch_bounds__(128)
attn_kernel(const bf16_t* __restrict__ xb,
            const bf16_t* __restrict__ xbT,
            bf16_t* __restrict__ attb) {
    const int wave = threadIdx.x >> 5;
    const int lane = threadIdx.x & 31;
    const int half = lane >> 4;       // which 16-lane group
    const int col  = lane & 15;       // query column owned by this lane
    const int qblocks = SEQ / 64;     // 64 q rows per block (4 waves x 16)
    const int b    = blockIdx.x / qblocks;
    const int qblk = blockIdx.x % qblocks;
    const int qbase = qblk * 64 + wave * 16;
    const int q = qbase + col;

    const bf16_t* xrow = xb  + (size_t)b * SEQ * DIM;
    const bf16_t* xT   = xbT + (size_t)b * DIM * SEQ;

    // Q^T B-operands, hoisted: one per 32-wide D chunk
    v16bf bq[4];
#pragma unroll
    for (int dch = 0; dch < 4; ++dch) {
        const bf16_t* qr = xrow + (size_t)q * DIM + dch * 32 + half * 8;
        bq[dch] = cat8(ld8(qr), ld8(qr + 16));
    }

    v8f acc[8];
    const v8f zero = {0.f, 0.f, 0.f, 0.f, 0.f, 0.f, 0.f, 0.f};
#pragma unroll
    for (int t = 0; t < 8; ++t) acc[t] = zero;

    float m_i = -__builtin_inff();
    float l_i = 0.f;

    // Interior steps: keys [kbase, kbase+32) entirely below the diagonal.
    int kbase = 0;
    for (; kbase + 32 <= qbase; kbase += 32) {
        // Pull next step's K rows and V^T lines toward the WGP while the
        // current WMMAs run (lowers to global_prefetch_b8; no counters).
        __builtin_prefetch(xrow + (size_t)(kbase + 32 + col) * DIM, 0, 1);
        __builtin_prefetch(xrow + (size_t)(kbase + 48 + col) * DIM, 0, 1);
        __builtin_prefetch(xT + (size_t)col * SEQ + kbase + 32, 0, 1);
        attn_step32<false>(xrow, xT, bq, acc, m_i, l_i, kbase, q, half, col);
    }
    // Exactly one diagonal (masked) step remains:
    //   qbase % 32 == 0  -> covers [qbase,   qbase+32), upper tile masked
    //   qbase % 32 == 16 -> covers [qbase-16, qbase+16)
    attn_step32<true>(xrow, xT, bq, acc, m_i, l_i, kbase, q, half, col);

    // ---- normalize, pack to bf16, store row-major att[b, q, :] ----
    const float linv = 1.f / l_i;
#pragma unroll
    for (int t = 0; t < 8; ++t) {
        v8bf pk;
#pragma unroll
        for (int v = 0; v < 8; ++v) pk[v] = (bf16_t)(acc[t][v] * linv);
        bf16_t* dst = attb + ((size_t)(b * SEQ + q) * DIM + t * 16 + half * 8);
        *reinterpret_cast<v8bf*>(dst) = pk;
    }
}

// ---------------------------------------------------------------------------
// Projection: out = [x, att] @ W^T + b   ([B*S,256] x [256,128])
// One wave = 16 output rows x 128 cols (8 accum tiles), K = 8 chunks of 32.
// ---------------------------------------------------------------------------
__global__ void __launch_bounds__(128)
proj_kernel(const bf16_t* __restrict__ xb,
            const bf16_t* __restrict__ attb,
            const bf16_t* __restrict__ Wb,
            const float*  __restrict__ bias,
            float* __restrict__ out) {
    const int wave = threadIdx.x >> 5;
    const int lane = threadIdx.x & 31;
    const int half = lane >> 4;
    const int col  = lane & 15;
    const int rowbase = blockIdx.x * 64 + wave * 16;   // rows into [B*S]

    v8f acc[8];
    const v8f zero = {0.f, 0.f, 0.f, 0.f, 0.f, 0.f, 0.f, 0.f};
#pragma unroll
    for (int t = 0; t < 8; ++t) acc[t] = zero;

#pragma unroll
    for (int kc = 0; kc < 8; ++kc) {
        const bf16_t* src = (kc < 4) ? xb : attb;
        const int dlo = (kc & 3) * 32 + half * 8;
        const bf16_t* ar = src + (size_t)(rowbase + col) * DIM + dlo;
        v16bf a = cat8(ld8(ar), ld8(ar + 16));
        const int klo = kc * 32 + half * 8;
#pragma unroll
        for (int nt = 0; nt < 8; ++nt) {
            const bf16_t* wr = Wb + (size_t)(nt * 16 + col) * (2 * DIM) + klo;
            v16bf bw = cat8(ld8(wr), ld8(wr + 16));
            acc[nt] = __builtin_amdgcn_wmma_f32_16x16x32_bf16(
                          false, a, false, bw, (short)0, acc[nt], false, false);
        }
    }

#pragma unroll
    for (int nt = 0; nt < 8; ++nt) {
        const float bv = bias[nt * 16 + col];
#pragma unroll
        for (int v = 0; v < 8; ++v) {
            out[(size_t)(rowbase + v + half * 8) * DIM + nt * 16 + col] =
                acc[nt][v] + bv;
        }
    }
}

// ---------------------------------------------------------------------------
extern "C" void kernel_launch(void* const* d_in, const int* in_sizes, int n_in,
                              void* d_out, int out_size, void* d_ws, size_t ws_size,
                              hipStream_t stream) {
    (void)in_sizes; (void)n_in; (void)out_size; (void)ws_size;
    const float* x    = (const float*)d_in[0];   // [B,S,D]
    const float* W    = (const float*)d_in[1];   // [D,2D]
    const float* bias = (const float*)d_in[2];   // [D]
    float* out = (float*)d_out;                  // [B,S,D]

    const size_t nx = (size_t)NB * SEQ * DIM;    // elements per bf16 tensor
    bf16_t* xb   = (bf16_t*)d_ws;
    bf16_t* xbT  = xb + nx;
    bf16_t* attb = xbT + nx;
    bf16_t* Wb   = attb + nx;                    // D*2D bf16

    cvt_x_kernel<<<(int)(nx / 256), 256, 0, stream>>>(x, xb, xbT);
    cvt_w_kernel<<<(DIM * 2 * DIM) / 256, 256, 0, stream>>>(W, Wb);

    attn_kernel<<<NB * (SEQ / 64), 128, 0, stream>>>(xb, xbT, attb);

    proj_kernel<<<(NB * SEQ) / 64, 128, 0, stream>>>(xb, attb, Wb, bias, out);
}
